// SelfAttention_53369263620701
// MI455X (gfx1250) — matrix operand order repaired and checked by reading, not compiled, not run on hardware
//
#include <hip/hip_runtime.h>

typedef __attribute__((ext_vector_type(16))) __bf16 v16bf;
typedef __attribute__((ext_vector_type(8)))  __bf16 v8bf;
typedef __attribute__((ext_vector_type(8)))  float  v8f;
typedef __attribute__((ext_vector_type(4)))  float  v4f;

namespace {

constexpr int kB = 4, kH = 16, kL = 2048, kD = 64;
constexpr float kScale = 0.125f;      // 1/sqrt(64)
constexpr float kNeg   = -1e9f;
constexpr int kPStride = 72;          // P row stride in bf16 (144B, 16B-aligned reads)

__device__ __forceinline__ v8f wmma_bf16(v16bf a, v16bf b, v8f c) {
  return __builtin_amdgcn_wmma_f32_16x16x32_bf16(false, a, false, b,
                                                 (short)0, c, false, false);
}

// 16 contiguous bf16 (32B, 16B-aligned) -> v16bf via two b128 loads.
__device__ __forceinline__ v16bf load16bf(const __bf16* p) {
  v8bf lo = *(const v8bf*)(p);
  v8bf hi = *(const v8bf*)(p + 8);
  return __builtin_shufflevector(lo, hi, 0, 1, 2, 3, 4, 5, 6, 7,
                                 8, 9, 10, 11, 12, 13, 14, 15);
}

// DPP16 lane exchange within 16-lane rows (VALU pipe; no LDS round trips).
template <int CTRL>
__device__ __forceinline__ float dpp_movf(float x) {
  return __builtin_bit_cast(
      float, __builtin_amdgcn_update_dpp(0, __builtin_bit_cast(int, x),
                                         CTRL, 0xF, 0xF, true));
}
// Butterfly over xor masks {1,2,7,15} == full 16-lane row reduction.
__device__ __forceinline__ float row_max16(float x) {
  x = fmaxf(x, dpp_movf<0xB1>(x));    // quad_perm(1,0,3,2)  : xor 1
  x = fmaxf(x, dpp_movf<0x4E>(x));    // quad_perm(2,3,0,1)  : xor 2
  x = fmaxf(x, dpp_movf<0x141>(x));   // row_half_mirror     : xor 7
  x = fmaxf(x, dpp_movf<0x140>(x));   // row_mirror          : xor 15
  return x;
}
__device__ __forceinline__ float row_sum16(float x) {
  x += dpp_movf<0xB1>(x);
  x += dpp_movf<0x4E>(x);
  x += dpp_movf<0x141>(x);
  x += dpp_movf<0x140>(x);
  return x;
}

// A-fragment (16x32 bf16) from an f32 row (one-time Q setup).
__device__ __forceinline__ v16bf load_a_frag_f32(const float* row, int lane) {
  const int k0 = (lane & 16) ? 16 : 0;
  v4f x0 = *(const v4f*)(row + k0);
  v4f x1 = *(const v4f*)(row + k0 + 4);
  v4f x2 = *(const v4f*)(row + k0 + 8);
  v4f x3 = *(const v4f*)(row + k0 + 12);
  v16bf a;
#pragma unroll
  for (int j = 0; j < 4; ++j) {
    a[j]      = (__bf16)x0[j];
    a[4 + j]  = (__bf16)x1[j];
    a[8 + j]  = (__bf16)x2[j];
    a[12 + j] = (__bf16)x3[j];
  }
  return a;
}

} // namespace

// ---------------------------------------------------------------------------
// Prep: K (f32) -> Kb (bf16, same layout); V (f32) -> Vt (bf16, B,H,D,L).
// ---------------------------------------------------------------------------
__global__ __launch_bounds__(256)
void prep_kv(const float* __restrict__ K, const float* __restrict__ V,
             __bf16* __restrict__ Kb, __bf16* __restrict__ Vt) {
  __shared__ __bf16 tile[64][72];
  const int bh = blockIdx.x >> 5;
  const int t  = blockIdx.x & 31;
  const size_t base = (size_t)bh * kL * kD;
  const int l0 = t * 64;
  const int r  = threadIdx.x >> 2;
  const int c  = (threadIdx.x & 3) * 16;

  {
    const float* src = K + base + (size_t)(l0 + r) * kD + c;
    __bf16* dst = Kb + base + (size_t)(l0 + r) * kD + c;
#pragma unroll
    for (int h = 0; h < 2; ++h) {
      v4f x0 = *(const v4f*)(src + 8 * h);
      v4f x1 = *(const v4f*)(src + 8 * h + 4);
      v8bf y;
#pragma unroll
      for (int j = 0; j < 4; ++j) { y[j] = (__bf16)x0[j]; y[4 + j] = (__bf16)x1[j]; }
      *(v8bf*)(dst + 8 * h) = y;
    }
  }
  {
    const float* src = V + base + (size_t)(l0 + r) * kD + c;
#pragma unroll
    for (int j = 0; j < 16; j += 4) {
      v4f x = *(const v4f*)(src + j);
      tile[r][c + j]     = (__bf16)x[0];
      tile[r][c + j + 1] = (__bf16)x[1];
      tile[r][c + j + 2] = (__bf16)x[2];
      tile[r][c + j + 3] = (__bf16)x[3];
    }
  }
  __syncthreads();
  {
    __bf16* dst = Vt + base + (size_t)r * kL + l0 + c;
#pragma unroll
    for (int h = 0; h < 2; ++h) {
      v8bf y;
#pragma unroll
      for (int j = 0; j < 8; ++j) y[j] = tile[c + 8 * h + j][r];
      *(v8bf*)(dst + 8 * h) = y;
    }
  }
}

// ---------------------------------------------------------------------------
// Flash attention forward: bf16 WMMA, fp32 softmax/accumulate.
// 1 wave = 16 query rows x 64-key tiles; 16 WMMAs per iteration.
// ---------------------------------------------------------------------------
__global__ __launch_bounds__(128)
void fa_fwd_bf16_wmma(const float* __restrict__ Q, const __bf16* __restrict__ Kb,
                      const __bf16* __restrict__ Vt, const int* __restrict__ Mask,
                      const float* __restrict__ Bias, float* __restrict__ Out) {
  __shared__ __bf16 ldsP[4][16 * kPStride];

  const int lane = threadIdx.x & 31;
  const int wid  = threadIdx.x >> 5;
  const int qt   = blockIdx.x & 31;
  const int bh   = blockIdx.x >> 5;
  const int b    = bh >> 4;
  const size_t base = (size_t)bh * kL * kD;
  const int q0 = qt * 64 + wid * 16;

  const int n     = lane & 15;
  const int half  = lane >> 4;
  const int khalf = half << 4;

  const float* qrow = Q + base + (size_t)(q0 + n) * kD;
  const v16bf aQ0 = load_a_frag_f32(qrow, lane);
  const v16bf aQ1 = load_a_frag_f32(qrow + 32, lane);

  v8f acc[4];
#pragma unroll
  for (int d = 0; d < 4; ++d) acc[d] = {};
  float mrow[8], lpart[8];                       // lpart: per-LANE partial sum
#pragma unroll
  for (int vg = 0; vg < 8; ++vg) { mrow[vg] = -3.0e38f; lpart[vg] = 0.0f; }

  const int*   mk    = Mask + (size_t)b * kL;
  const float* biasB = Bias + (size_t)b * kL * kL + (size_t)(q0 + (half << 3)) * kL;
  __bf16* lp = ldsP[wid];

  for (int kv = 0; kv < kL; kv += 64) {
    // ---- S = Q K^T for 64 keys: 4 C-frags x 2 contraction chunks ----
    const __bf16* krow = Kb + base + (size_t)(kv + n) * kD + khalf;
    v8f s[4];
#pragma unroll
    for (int g = 0; g < 4; ++g) {
      const __bf16* kr = krow + (size_t)(16 * g) * kD;
      v8f t = {};
      t = wmma_bf16(aQ0, load16bf(kr), t);
      t = wmma_bf16(aQ1, load16bf(kr + 32), t);
      s[g] = t;
    }
    if (kv + 64 < kL) {
#pragma unroll
      for (int g = 0; g < 4; ++g)
        __builtin_prefetch(krow + (size_t)(64 + 16 * g) * kD, 0, 1);
    }

    // Unconditional bias/mask loads (coalesced clauses; no exec branches).
    bool okg[4];
#pragma unroll
    for (int g = 0; g < 4; ++g) okg[g] = mk[kv + 16 * g + n] != 0;
    float bb[4][8];
#pragma unroll
    for (int vg = 0; vg < 8; ++vg) {
      const float* brow = biasB + (size_t)vg * kL + kv + n;
#pragma unroll
      for (int g = 0; g < 4; ++g) bb[g][vg] = brow[16 * g];
    }

    // ---- online softmax: one DPP row-max + one corr per 64 keys ----
    float corr[8];
#pragma unroll
    for (int vg = 0; vg < 8; ++vg) {
      float v[4];
#pragma unroll
      for (int g = 0; g < 4; ++g) {
        const float x = fmaf(s[g][vg], kScale, bb[g][vg]);
        v[g] = okg[g] ? x : kNeg;                // v_cndmask
      }
      float mx = fmaxf(fmaxf(v[0], v[1]), fmaxf(v[2], v[3]));
      mx = row_max16(mx);
      const float mnew = fmaxf(mrow[vg], mx);
      corr[vg] = __expf(mrow[vg] - mnew);
      mrow[vg] = mnew;
      float e[4];
#pragma unroll
      for (int g = 0; g < 4; ++g) e[g] = __expf(v[g] - mnew);
      lpart[vg] = fmaf(lpart[vg], corr[vg], (e[0] + e[1]) + (e[2] + e[3]));
      const int row = vg + (half << 3);
#pragma unroll
      for (int g = 0; g < 4; ++g)
        lp[row * kPStride + 16 * g + n] = (__bf16)e[g];
    }

#pragma unroll
    for (int d = 0; d < 4; ++d)
#pragma unroll
      for (int vg = 0; vg < 8; ++vg) acc[d][vg] *= corr[vg];

    asm volatile("s_wait_dscnt 0x0" ::: "memory");

    // P back in A-layout: two 16x32 fragments, four 16B ds reads.
    v16bf aP0, aP1;
    {
      const __bf16* r = lp + n * kPStride;
      const int c0 = (lane & 16) ? 16 : 0;
      aP0 = __builtin_shufflevector(*(const v8bf*)(r + c0),
                                    *(const v8bf*)(r + c0 + 8),
                                    0, 1, 2, 3, 4, 5, 6, 7,
                                    8, 9, 10, 11, 12, 13, 14, 15);
      aP1 = __builtin_shufflevector(*(const v8bf*)(r + c0 + 32),
                                    *(const v8bf*)(r + c0 + 40),
                                    0, 1, 2, 3, 4, 5, 6, 7,
                                    8, 9, 10, 11, 12, 13, 14, 15);
    }

    // ---- O += P @ V: 8 WMMAs (2 key chunks x 4 d-blocks) ----
    const __bf16* vt = Vt + base + (size_t)n * kL + kv + khalf;
#pragma unroll
    for (int d = 0; d < 4; ++d) {
      const __bf16* vd = vt + (size_t)(d * 16) * kL;
      v8f t = acc[d];
      t = wmma_bf16(aP0, load16bf(vd), t);
      t = wmma_bf16(aP1, load16bf(vd + 32), t);
      acc[d] = t;
    }
    if (kv + 64 < kL) __builtin_prefetch(vt + 64, 0, 1);
  }

  // ---- epilogue: one row-sum reduction, then scale + store ----
#pragma unroll
  for (int vg = 0; vg < 8; ++vg) lpart[vg] = 1.0f / row_sum16(lpart[vg]);
#pragma unroll
  for (int d = 0; d < 4; ++d)
#pragma unroll
    for (int vg = 0; vg < 8; ++vg)
      __builtin_nontemporal_store(
          acc[d][vg] * lpart[vg],
          Out + base + (size_t)(q0 + vg + (half << 3)) * kD + d * 16 + n);
}

extern "C" void kernel_launch(void* const* d_in, const int* in_sizes, int n_in,
                              void* d_out, int out_size, void* d_ws, size_t ws_size,
                              hipStream_t stream) {
  (void)in_sizes; (void)n_in; (void)ws_size; (void)out_size;
  const float* Q    = (const float*)d_in[0];
  const float* K    = (const float*)d_in[1];
  const float* V    = (const float*)d_in[2];
  const int*   Mask = (const int*)d_in[3];
  const float* Bias = (const float*)d_in[4];
  float* Out = (float*)d_out;

  __bf16* Kb = (__bf16*)d_ws;
  __bf16* Vt = Kb + (size_t)kB * kH * kL * kD;

  dim3 pgrid(kB * kH * (kL / 64));
  hipLaunchKernelGGL(prep_kv, pgrid, dim3(256), 0, stream, K, V, Kb, Vt);

  dim3 grid(kB * kH * (kL / 64));
  hipLaunchKernelGGL(fa_fwd_bf16_wmma, grid, dim3(128), 0, stream,
                     Q, Kb, Vt, Mask, Bias, Out);
}
